// LSTMModel_49151605736134
// MI455X (gfx1250) — matrix operand order repaired and checked by reading, not compiled, not run on hardware
//
#include <hip/hip_runtime.h>

typedef __bf16 bf16;
typedef __attribute__((ext_vector_type(16))) __bf16 v16bf;
typedef __attribute__((ext_vector_type(8)))  __bf16 v8bf;
typedef __attribute__((ext_vector_type(8)))  float  v8f;
typedef __attribute__((ext_vector_type(4)))  float  v4f;

#define B_   2048
#define T_   512
#define F_   32
#define H_   128
#define HPAD 136   // LDS row stride in bf16 elements (bank-conflict padding)

// ---------------------------------------------------------------------------
// helpers
// ---------------------------------------------------------------------------
__device__ __forceinline__ float fast_sigmoid(float x) {
  return __builtin_amdgcn_rcpf(1.0f + __expf(-x));
}
__device__ __forceinline__ float fast_tanh(float x) {
  // tanh(x) = 1 - 2/(exp(2x)+1); saturates correctly at +/-1
  return 1.0f - 2.0f * __builtin_amdgcn_rcpf(1.0f + __expf(2.0f * x));
}
__device__ __forceinline__ v8f splat8(float v) {
  v8f r;
#pragma unroll
  for (int i = 0; i < 8; ++i) r[i] = v;
  return r;
}

#define WMMA_BF16(Af, Bf, Cf) \
  __builtin_amdgcn_wmma_f32_16x16x32_bf16(false, (Af), false, (Bf), (short)0, (Cf), false, false)

// B-operand fragment: fragment-ordered global buffer, 32 bytes per lane, coalesced.
__device__ __forceinline__ v16bf ldB(const bf16* wf, int nt, int KB, int kb, int lane) {
  return *(const v16bf*)(wf + ((size_t)((nt * KB) + kb) * 32 + lane) * 16);
}

// A-operand fragment (16x32 bf16) from row-major LDS h buffer.
// lane<16 : row=lane,    elems = K[kb*32+0..7],  K[kb*32+16..23]
// lane>=16: row=lane-16, elems = K[kb*32+8..15], K[kb*32+24..31]
__device__ __forceinline__ v16bf ldA_lds(const bf16* row, int kb, int half) {
  const bf16* p = row + kb * 32 + half * 8;
  v8bf lo = *(const v8bf*)(p);
  v8bf hi = *(const v8bf*)(p + 16);
  return __builtin_shufflevector(lo, hi, 0,1,2,3,4,5,6,7,8,9,10,11,12,13,14,15);
}

// ---------------------------------------------------------------------------
// weight repack: W (4H x K) fp32 row-major -> bf16 B-operand fragments
// frag element index: (((nt*KB + kb)*32 + lane)*16 + e)
//   n = nt*16 + (lane&15);  k = kb*32 + (e>=8 ? 16 : 0) + (lane>=16 ? 8 : 0) + (e&7);  val = W[n*K + k]
// ---------------------------------------------------------------------------
__global__ void repack_w(const float* __restrict__ W, bf16* __restrict__ dst,
                         int K, int total) {
  int idx = blockIdx.x * 256 + threadIdx.x;
  if (idx >= total) return;
  int e    = idx & 15;
  int lane = (idx >> 4) & 31;
  int fidx = idx >> 9;           // nt*KB + kb
  int KB   = K >> 5;
  int kb   = fidx % KB;
  int nt   = fidx / KB;
  int n    = nt * 16 + (lane & 15);
  int k    = kb * 32 + ((e >> 3) << 4) + ((lane >> 4) << 3) + (e & 7);
  dst[idx] = (bf16)W[n * K + k];
}

// ---------------------------------------------------------------------------
// fused 2-layer LSTM + FC. One WG = 16 batch rows, 8 waves, all T steps.
// Wave w owns hidden units [16w, 16w+16): gate tiles nt = gate*8 + w.
// Whh0/Whh1 fragments live in registers; Wih0/Wih1 stream from L2 each step.
// ---------------------------------------------------------------------------
__global__ __launch_bounds__(256)
void lstm_fused(const float* __restrict__ x,
                const float* __restrict__ b0, const float* __restrict__ b1,
                const bf16* __restrict__ wih0f, const bf16* __restrict__ whh0f,
                const bf16* __restrict__ wih1f, const bf16* __restrict__ whh1f,
                const float* __restrict__ Wfc,  const float* __restrict__ bfc,
                float* __restrict__ out) {
  __shared__ __align__(16) bf16 h0buf[2][16][HPAD];
  __shared__ __align__(16) bf16 h1buf[2][16][HPAD];

  const int tid  = threadIdx.x;
  const int w    = tid >> 5;    // wave id 0..7
  const int lane = tid & 31;
  const int half = lane >> 4;
  const int hl   = lane & 15;
  const int r0   = blockIdx.x * 16;

  // zero the t=0 state buffers
  for (int i = tid; i < 16 * HPAD; i += 256) {
    (&h0buf[0][0][0])[i] = (bf16)0.0f;
    (&h1buf[0][0][0])[i] = (bf16)0.0f;
  }

  // biases: accumulator column N == hl for every element in this lane
  float bias0[4], bias1[4];
#pragma unroll
  for (int g = 0; g < 4; ++g) {
    bias0[g] = b0[g * H_ + w * 16 + hl];
    bias1[g] = b1[g * H_ + w * 16 + hl];
  }

  // ---- preload recurrent weights into registers: 32 frags = 256 VGPRs ----
  v16bf rWhh0[4][4], rWhh1[4][4];
#pragma unroll
  for (int g = 0; g < 4; ++g)
#pragma unroll
    for (int kb = 0; kb < 4; ++kb) {
      rWhh0[g][kb] = ldB(whh0f, g * 8 + w, 4, kb, lane);
      rWhh1[g][kb] = ldB(whh1f, g * 8 + w, 4, kb, lane);
    }

  v8f c0 = splat8(0.0f), c1 = splat8(0.0f);
  int cur = 0;
  __syncthreads();

  for (int t = 0; t < T_; ++t) {
    const int nxt = cur ^ 1;

    // Launder a zero offset so the streamed Wih loads cannot be hoisted
    // out of the loop (prevents the 400-VGPR hoist + scratch spills).
    int zoff = 0;
    asm volatile("" : "+v"(zoff));
    const bf16* wi0 = wih0f + zoff;
    const bf16* wi1 = wih1f + zoff;

    // ---- x A-fragment (16x32) straight from global fp32, convert to bf16 ----
    const float* px = x + ((size_t)(r0 + hl) * T_ + t) * F_ + half * 8;
    v4f q[4];
    q[0] = *(const v4f*)(px);
    q[1] = *(const v4f*)(px + 4);
    q[2] = *(const v4f*)(px + 16);
    q[3] = *(const v4f*)(px + 20);
    v16bf ax;
#pragma unroll
    for (int j = 0; j < 4; ++j)
#pragma unroll
      for (int r = 0; r < 4; ++r) ax[j * 4 + r] = (bf16)q[j][r];

    // ---- layer 0: z = x_t @ Wih0^T + h0 @ Whh0^T + b0 ----
    v8f acc[4];
#pragma unroll
    for (int g = 0; g < 4; ++g) {
      acc[g] = splat8(bias0[g]);
      acc[g] = WMMA_BF16(ax, ldB(wi0, g * 8 + w, 1, 0, lane), acc[g]);
    }
#pragma unroll
    for (int kb = 0; kb < 4; ++kb) {
      v16bf a = ldA_lds(&h0buf[cur][hl][0], kb, half);
#pragma unroll
      for (int g = 0; g < 4; ++g)
        acc[g] = WMMA_BF16(a, rWhh0[g][kb], acc[g]);
    }

    v8f hv0;
#pragma unroll
    for (int r = 0; r < 8; ++r) {
      float iv = fast_sigmoid(acc[0][r]);
      float fv = fast_sigmoid(acc[1][r]);
      float gv = fast_tanh(acc[2][r]);
      float ov = fast_sigmoid(acc[3][r]);
      float cv = fv * c0[r] + iv * gv;
      c0[r]  = cv;
      hv0[r] = ov * fast_tanh(cv);
    }
    // publish h0(t): element r of accumulator layout is row (r + 8*half), col hl
#pragma unroll
    for (int r = 0; r < 8; ++r)
      h0buf[nxt][r + 8 * half][w * 16 + hl] = (bf16)hv0[r];
    __syncthreads();

    // ---- layer 1: z = h0(t) @ Wih1^T + h1(t-1) @ Whh1^T + b1 ----
#pragma unroll
    for (int g = 0; g < 4; ++g) acc[g] = splat8(bias1[g]);
#pragma unroll
    for (int kb = 0; kb < 4; ++kb) {
      v16bf a = ldA_lds(&h0buf[nxt][hl][0], kb, half);
#pragma unroll
      for (int g = 0; g < 4; ++g)
        acc[g] = WMMA_BF16(a, ldB(wi1, g * 8 + w, 4, kb, lane), acc[g]);
    }
#pragma unroll
    for (int kb = 0; kb < 4; ++kb) {
      v16bf a = ldA_lds(&h1buf[cur][hl][0], kb, half);
#pragma unroll
      for (int g = 0; g < 4; ++g)
        acc[g] = WMMA_BF16(a, rWhh1[g][kb], acc[g]);
    }

    v8f hv1;
#pragma unroll
    for (int r = 0; r < 8; ++r) {
      float iv = fast_sigmoid(acc[0][r]);
      float fv = fast_sigmoid(acc[1][r]);
      float gv = fast_tanh(acc[2][r]);
      float ov = fast_sigmoid(acc[3][r]);
      float cv = fv * c1[r] + iv * gv;
      c1[r]  = cv;
      hv1[r] = ov * fast_tanh(cv);
    }
#pragma unroll
    for (int r = 0; r < 8; ++r)
      h1buf[nxt][r + 8 * half][w * 16 + hl] = (bf16)hv1[r];

    cur = nxt;
    __syncthreads();
  }

  // ---- final FC: out[b] = h1_last[b,:] . Wfc + bfc ----
  if (tid < 16) {
    float s = bfc[0];
#pragma unroll 8
    for (int u = 0; u < H_; ++u)
      s += (float)h1buf[cur][tid][u] * Wfc[u];
    out[r0 + tid] = s;
  }
}

// ---------------------------------------------------------------------------
// launch
// ---------------------------------------------------------------------------
extern "C" void kernel_launch(void* const* d_in, const int* in_sizes, int n_in,
                              void* d_out, int out_size, void* d_ws, size_t ws_size,
                              hipStream_t stream) {
  (void)in_sizes; (void)n_in; (void)out_size; (void)ws_size;
  const float* x    = (const float*)d_in[0];
  const float* Wih0 = (const float*)d_in[1];
  const float* Whh0 = (const float*)d_in[2];
  const float* b0   = (const float*)d_in[3];
  const float* Wih1 = (const float*)d_in[4];
  const float* Whh1 = (const float*)d_in[5];
  const float* b1   = (const float*)d_in[6];
  const float* Wfc  = (const float*)d_in[7];
  const float* bfc  = (const float*)d_in[8];
  float* out = (float*)d_out;

  char* ws = (char*)d_ws;
  bf16* wih0f = (bf16*)(ws);                      // 32*1*512 el = 32 KB
  bf16* whh0f = (bf16*)(ws + 32768);              // 32*4*512 el = 128 KB
  bf16* wih1f = (bf16*)(ws + 32768 + 131072);     // 128 KB
  bf16* whh1f = (bf16*)(ws + 32768 + 2 * 131072); // 128 KB

  repack_w<<<(16384 + 255) / 256, 256, 0, stream>>>(Wih0, wih0f, 32, 16384);
  repack_w<<<(65536 + 255) / 256, 256, 0, stream>>>(Whh0, whh0f, 128, 65536);
  repack_w<<<(65536 + 255) / 256, 256, 0, stream>>>(Wih1, wih1f, 128, 65536);
  repack_w<<<(65536 + 255) / 256, 256, 0, stream>>>(Whh1, whh1f, 128, 65536);

  lstm_fused<<<B_ / 16, 256, 0, stream>>>(x, b0, b1, wih0f, whh0f, wih1f, whh1f,
                                          Wfc, bfc, out);
}
